// MogLSTM_31585189494947
// MI455X (gfx1250) — matrix operand order repaired and checked by reading, not compiled, not run on hardware
//
#include <hip/hip_runtime.h>

typedef __attribute__((ext_vector_type(16))) _Float16 v16h;
typedef __attribute__((ext_vector_type(8)))  _Float16 v8h;
typedef __attribute__((ext_vector_type(8)))  float    v8f;

#define SEQ 256
#define BAT 64
#define HID 512
#define VOC 10000

__device__ __forceinline__ v8f wmma16(v16h a, v16h b, v8f c) {
  return __builtin_amdgcn_wmma_f32_16x16x32_f16(false, a, false, b, (short)0, c, false, false);
}

// A-matrix 16x32 f16 fragment: lane m=l&15, kh=l>>4.
// VGPR0..3 <- K = k0+8*kh .. +7 ; VGPR4..7 <- K = k0+16+8*kh .. +7
__device__ __forceinline__ v16h ld_a_frag(const _Float16* row, int k0, int kh) {
  const v8h lo = *(const v8h*)(row + k0 + 8 * kh);
  const v8h hi = *(const v8h*)(row + k0 + 8 * kh + 16);
  v16h r;
#pragma unroll
  for (int i = 0; i < 8; ++i) { r[i] = lo[i]; r[i + 8] = hi[i]; }
  return r;
}

// B-matrix 32x16 f16 fragment: lane n=l&15, kh=l>>4.
// VGPR0..3 <- K = k0+8*kh .. +7 ; VGPR4..7 <- K = k0+16+8*kh .. +7
__device__ __forceinline__ v16h ld_b_frag(const _Float16* row, int k0, int kh) {
  const v8h lo = *(const v8h*)(row + k0 + 8 * kh);
  const v8h hi = *(const v8h*)(row + k0 + 8 * kh + 16);
  v16h r;
#pragma unroll
  for (int i = 0; i < 8; ++i) { r[i] = lo[i]; r[i + 8] = hi[i]; }
  return r;
}

__device__ __forceinline__ float sigm(float x) { return 1.0f / (1.0f + __expf(-x)); }

// ---------------- utility kernels ----------------
__global__ void k_cvt_f16(const float* __restrict__ s, _Float16* __restrict__ d, int n) {
  int i = blockIdx.x * blockDim.x + threadIdx.x;
  if (i < n) d[i] = (_Float16)s[i];
}

__global__ void k_init_state(const float* __restrict__ h1, const float* __restrict__ c1,
                             _Float16* __restrict__ h1_16, float* __restrict__ c1w,
                             _Float16* __restrict__ h2_16, float* __restrict__ c2w, int n) {
  int i = blockIdx.x * blockDim.x + threadIdx.x;
  if (i < n) {
    h1_16[i] = (_Float16)h1[i];
    c1w[i]   = c1[i];
    h2_16[i] = (_Float16)0.0f;
    c2w[i]   = 0.0f;
  }
}

// ---------------- mogrifier stage ----------------
// T[m,n] = 2*sigmoid( sum_k A[m,k]*W[n,k] + bias[n] ) * T[m,n]
// A,T: [64,512] f16 ; W: [512,512] f16 ; bias: [512] f32
// grid: 8 blocks (n0 = blockIdx.x*64), block: 256 thr = 8 waves
__global__ __launch_bounds__(256) void k_mog(const _Float16* __restrict__ A,
                                             const _Float16* __restrict__ W,
                                             const float* __restrict__ bias,
                                             _Float16* __restrict__ T) {
  const int lane = threadIdx.x & 31;
  const int wave = threadIdx.x >> 5;
  const int m0 = (wave & 3) * 16;
  const int n0 = blockIdx.x * 64 + (wave >> 2) * 32;
  const int ml = lane & 15, kh = lane >> 4;

  const _Float16* arow = A + (size_t)(m0 + ml) * HID;
  const _Float16* brow0 = W + (size_t)(n0 + ml) * HID;
  const _Float16* brow1 = W + (size_t)(n0 + 16 + ml) * HID;

  v8f acc0 = {}; v8f acc1 = {};
#pragma unroll 4
  for (int k = 0; k < HID; k += 32) {
    v16h a  = ld_a_frag(arow, k, kh);
    v16h b0 = ld_b_frag(brow0, k, kh);
    v16h b1 = ld_b_frag(brow1, k, kh);
    acc0 = wmma16(a, b0, acc0);
    acc1 = wmma16(a, b1, acc1);
  }
  const int nl = n0 + (lane & 15);
  const float bv0 = bias[nl], bv1 = bias[nl + 16];
#pragma unroll
  for (int r = 0; r < 8; ++r) {
    const int m = m0 + r + 8 * kh;
    _Float16* t = T + (size_t)m * HID + nl;
    float u0 = 2.0f * sigm(acc0[r] + bv0);
    t[0]  = (_Float16)(u0 * (float)t[0]);
    float u1 = 2.0f * sigm(acc1[r] + bv1);
    t[16] = (_Float16)(u1 * (float)t[16]);
  }
}

// ---------------- fused LSTM gates + pointwise ----------------
// gates = X@Wih^T + Hm@Whh^T + bih + bhh  (order i,f,g,o); then c/h update.
// grid: 4 blocks (128 hidden cols each), block: 512 thr = 16 waves
__global__ __launch_bounds__(512) void k_gates(const _Float16* __restrict__ X,
                                               const _Float16* __restrict__ Hm,
                                               const _Float16* __restrict__ Wih,
                                               const _Float16* __restrict__ Whh,
                                               const float* __restrict__ bih,
                                               const float* __restrict__ bhh,
                                               float* __restrict__ C,
                                               _Float16* __restrict__ Hout,
                                               _Float16* __restrict__ Hcopy,
                                               _Float16* __restrict__ Hs16,
                                               float* __restrict__ Hs32) {
  const int lane = threadIdx.x & 31;
  const int wave = threadIdx.x >> 5;
  const int m0 = (wave & 3) * 16;
  const int nb = blockIdx.x * 128 + (wave >> 2) * 32;
  const int ml = lane & 15, kh = lane >> 4;

  const _Float16* xrow = X  + (size_t)(m0 + ml) * HID;
  const _Float16* hrow = Hm + (size_t)(m0 + ml) * HID;

  v8f acc[4][2];
#pragma unroll
  for (int g = 0; g < 4; ++g) { acc[g][0] = (v8f){}; acc[g][1] = (v8f){}; }

  for (int k = 0; k < HID; k += 32) {
    v16h ax = ld_a_frag(xrow, k, kh);
    v16h ah = ld_a_frag(hrow, k, kh);
#pragma unroll
    for (int g = 0; g < 4; ++g) {
      const size_t rb = (size_t)(g * HID + nb + ml) * HID;
      v16h b;
      b = ld_b_frag(Wih + rb, k, kh);              acc[g][0] = wmma16(ax, b, acc[g][0]);
      b = ld_b_frag(Wih + rb + 16 * HID, k, kh);   acc[g][1] = wmma16(ax, b, acc[g][1]);
      b = ld_b_frag(Whh + rb, k, kh);              acc[g][0] = wmma16(ah, b, acc[g][0]);
      b = ld_b_frag(Whh + rb + 16 * HID, k, kh);   acc[g][1] = wmma16(ah, b, acc[g][1]);
    }
  }

  const int nl = nb + (lane & 15);
#pragma unroll
  for (int t = 0; t < 2; ++t) {
    const int n = nl + t * 16;
    const float bi = bih[n]            + bhh[n];
    const float bf = bih[HID + n]      + bhh[HID + n];
    const float bg = bih[2 * HID + n]  + bhh[2 * HID + n];
    const float bo = bih[3 * HID + n]  + bhh[3 * HID + n];
#pragma unroll
    for (int r = 0; r < 8; ++r) {
      const int m = m0 + r + 8 * kh;
      const size_t idx = (size_t)m * HID + n;
      const float ig = acc[0][t][r] + bi;
      const float fg = acc[1][t][r] + bf;
      const float gg = acc[2][t][r] + bg;
      const float og = acc[3][t][r] + bo;
      const float cn = sigm(fg) * C[idx] + sigm(ig) * tanhf(gg);
      C[idx] = cn;
      const float hn = sigm(og) * tanhf(cn);
      const _Float16 hh = (_Float16)hn;
      Hout[idx] = hh;
      if (Hcopy) Hcopy[idx] = hh;
      if (Hs16)  Hs16[idx]  = hh;
      if (Hs32)  Hs32[idx]  = hn;
    }
  }
}

// ---------------- batched output projection ----------------
// Out[M=16384, N=10000] = Hs16[M,512] @ fcW16[N,512]^T + fcb
// grid: (256, 79), block: 512 thr = 16 waves; block tile 64x128, wave tile 16x32
__global__ __launch_bounds__(512) void k_fc(const _Float16* __restrict__ A,
                                            const _Float16* __restrict__ W,
                                            const float* __restrict__ bias,
                                            float* __restrict__ Out) {
  const int lane = threadIdx.x & 31;
  const int wave = threadIdx.x >> 5;
  const int m0 = blockIdx.x * 64 + (wave & 3) * 16;
  const int n0 = blockIdx.y * 128 + (wave >> 2) * 32;
  const int ml = lane & 15, kh = lane >> 4;
  const bool v0 = (n0 + 16) <= VOC;      // VOC % 16 == 0 -> subtile all-or-nothing
  const bool v1 = (n0 + 32) <= VOC;

  const _Float16* arow  = A + (size_t)(m0 + ml) * HID;
  const _Float16* brow0 = W + (size_t)(n0 + ml) * HID;
  const _Float16* brow1 = W + (size_t)(n0 + 16 + ml) * HID;

  v8f acc0 = {}; v8f acc1 = {};
#pragma unroll 4
  for (int k = 0; k < HID; k += 32) {
    v16h a = ld_a_frag(arow, k, kh);
    if (v0) { v16h b = ld_b_frag(brow0, k, kh); acc0 = wmma16(a, b, acc0); }
    if (v1) { v16h b = ld_b_frag(brow1, k, kh); acc1 = wmma16(a, b, acc1); }
  }
  const int nl = n0 + (lane & 15);
#pragma unroll
  for (int r = 0; r < 8; ++r) {
    const int m = m0 + r + 8 * kh;
    float* orow = Out + (size_t)m * VOC;
    if (v0) orow[nl]      = acc0[r] + bias[nl];
    if (v1) orow[nl + 16] = acc1[r] + bias[nl + 16];
  }
}

// ---------------- host launch ----------------
extern "C" void kernel_launch(void* const* d_in, const int* in_sizes, int n_in,
                              void* d_out, int out_size, void* d_ws, size_t ws_size,
                              hipStream_t stream) {
  (void)in_sizes; (void)n_in; (void)out_size; (void)ws_size;
  // Inputs (setup_inputs insertion order, params dicts flattened in insertion order)
  const float* input_seq = (const float*)d_in[0];
  const float* h1_in = (const float*)d_in[1];
  const float* c1_in = (const float*)d_in[2];
  const float *mogW1[5], *mogb1[5], *mogW2[5], *mogb2[5];
  for (int i = 0; i < 5; ++i) { mogW1[i] = (const float*)d_in[3 + i];  mogb1[i] = (const float*)d_in[8 + i]; }
  const float* Wih1 = (const float*)d_in[13]; const float* Whh1 = (const float*)d_in[14];
  const float* bih1 = (const float*)d_in[15]; const float* bhh1 = (const float*)d_in[16];
  for (int i = 0; i < 5; ++i) { mogW2[i] = (const float*)d_in[17 + i]; mogb2[i] = (const float*)d_in[22 + i]; }
  const float* Wih2 = (const float*)d_in[27]; const float* Whh2 = (const float*)d_in[28];
  const float* bih2 = (const float*)d_in[29]; const float* bhh2 = (const float*)d_in[30];
  const float* fcW  = (const float*)d_in[31]; const float* fcb  = (const float*)d_in[32];

  float* out_logits = (float*)d_out;                       // [S,B,V]
  float* out_hidden = out_logits + (size_t)SEQ * BAT * VOC; // [S,B,H]

  // Workspace carve (16B+ aligned chunks)
  char* ws = (char*)d_ws;
  auto take = [&](size_t bytes) -> char* {
    char* p = ws; ws += (bytes + 255) & ~(size_t)255; return p;
  };
  const size_t WSQ = (size_t)HID * HID;           // 512*512
  _Float16* mw1[5]; for (int i = 0; i < 5; ++i) mw1[i] = (_Float16*)take(WSQ * 2);
  _Float16* mw2[5]; for (int i = 0; i < 5; ++i) mw2[i] = (_Float16*)take(WSQ * 2);
  _Float16* wih1_16 = (_Float16*)take(4 * WSQ * 2);
  _Float16* whh1_16 = (_Float16*)take(4 * WSQ * 2);
  _Float16* wih2_16 = (_Float16*)take(4 * WSQ * 2);
  _Float16* whh2_16 = (_Float16*)take(4 * WSQ * 2);
  _Float16* fcw16   = (_Float16*)take((size_t)VOC * HID * 2);
  _Float16* hs16    = (_Float16*)take((size_t)SEQ * BAT * HID * 2);
  _Float16* x1      = (_Float16*)take((size_t)BAT * HID * 2);
  _Float16* x2      = (_Float16*)take((size_t)BAT * HID * 2);
  _Float16* h1b[2]  = { (_Float16*)take((size_t)BAT * HID * 2), (_Float16*)take((size_t)BAT * HID * 2) };
  _Float16* h2b[2]  = { (_Float16*)take((size_t)BAT * HID * 2), (_Float16*)take((size_t)BAT * HID * 2) };
  float*    c1w     = (float*)take((size_t)BAT * HID * 4);
  float*    c2w     = (float*)take((size_t)BAT * HID * 4);

  auto cvt = [&](const float* s, _Float16* d, size_t n) {
    k_cvt_f16<<<dim3((unsigned)((n + 255) / 256)), 256, 0, stream>>>(s, d, (int)n);
  };
  for (int i = 0; i < 5; ++i) cvt(mogW1[i], mw1[i], WSQ);
  for (int i = 0; i < 5; ++i) cvt(mogW2[i], mw2[i], WSQ);
  cvt(Wih1, wih1_16, 4 * WSQ); cvt(Whh1, whh1_16, 4 * WSQ);
  cvt(Wih2, wih2_16, 4 * WSQ); cvt(Whh2, whh2_16, 4 * WSQ);
  cvt(fcW, fcw16, (size_t)VOC * HID);

  k_init_state<<<dim3(BAT * HID / 256), 256, 0, stream>>>(h1_in, c1_in, h1b[0], c1w, h2b[0], c2w, BAT * HID);

  int cur = 0;
  for (int s = 0; s < SEQ; ++s) {
    const int nxt = cur ^ 1;
    // x_t -> f16
    cvt(input_seq + (size_t)s * BAT * HID, x1, (size_t)BAT * HID);
    // ---- cell 1 ----
    for (int i = 0; i < 5; ++i) {
      const _Float16* A = (i & 1) ? x1 : h1b[cur];
      _Float16*       T = (i & 1) ? h1b[cur] : x1;
      k_mog<<<dim3(8), 256, 0, stream>>>(A, mw1[i], mogb1[i], T);
    }
    k_gates<<<dim3(4), 512, 0, stream>>>(x1, h1b[cur], wih1_16, whh1_16, bih1, bhh1,
                                         c1w, h1b[nxt], /*Hcopy=*/x2, nullptr, nullptr);
    // ---- cell 2 ----
    for (int i = 0; i < 5; ++i) {
      const _Float16* A = (i & 1) ? x2 : h2b[cur];
      _Float16*       T = (i & 1) ? h2b[cur] : x2;
      k_mog<<<dim3(8), 256, 0, stream>>>(A, mw2[i], mogb2[i], T);
    }
    k_gates<<<dim3(4), 512, 0, stream>>>(x2, h2b[cur], wih2_16, whh2_16, bih2, bhh2,
                                         c2w, h2b[nxt], nullptr,
                                         hs16 + (size_t)s * BAT * HID,
                                         out_hidden + (size_t)s * BAT * HID);
    cur = nxt;
  }

  // batched projection: [S*B, H] x [H, V]
  k_fc<<<dim3(SEQ * BAT / 64, (VOC + 127) / 128), 512, 0, stream>>>(hs16, fcw16, fcb, out_logits);
}
